// RETRO_21895743275293
// MI455X (gfx1250) — matrix operand order repaired and checked by reading, not compiled, not run on hardware
//
#include <hip/hip_runtime.h>

// ---------------- model constants ----------------
#define BB      2
#define NN      512
#define KCH     8
#define RR      2
#define CNN     64
#define VOCAB   32000
#define ENCD    896
#define DECD    768
#define NHEADS  8
#define DHEAD   64
#define INNER   512

typedef __attribute__((ext_vector_type(16))) _Float16 v16h;
typedef __attribute__((ext_vector_type(8)))  _Float16 v8h;
typedef __attribute__((ext_vector_type(8)))  float    v8f;

// =====================================================================
// Common GEMM semantics: C[M,N] = act(alpha * A[M,K] @ B + bias) + resid
//   A: fp32 row-major M x K
//   B: fp32, opB==0 -> K x N row-major ; opB==1 -> N x K row-major
// Two tile variants:
//   gemm_wmma64 : 64x64 tile, 128 thr (4 waves, 32x32/wave), BK=64
//   gemm_wmma128: 128x128 tile, 256 thr (8 waves, 32x64/wave), BK=32
// Both: fp32->f16 convert during register-staged prefetch, double-buffered
// LDS, one barrier per K-step, v_wmma_f32_16x16x32_f16 core.
// =====================================================================

// A frag (16x32 f16): lane m=lane&15, h=lane>>4 holds K = h*8..+7 and 16+h*8..+7
__device__ __forceinline__ v16h ldfragA(const _Float16* p, int fh) {
    v8h lo = *(const v8h*)(p + fh * 8);
    v8h hi = *(const v8h*)(p + 16 + fh * 8);
    return __builtin_shufflevector(lo, hi, 0,1,2,3,4,5,6,7,8,9,10,11,12,13,14,15);
}
// B frag (32x16 f16): lane n=lane&15, h=lane>>4 holds K = h*8..+7 and 16+h*8..+7
__device__ __forceinline__ v16h ldfragB(const _Float16* p, int fh) {
    v8h lo = *(const v8h*)(p + fh * 8);
    v8h hi = *(const v8h*)(p + 16 + fh * 8);
    return __builtin_shufflevector(lo, hi, 0,1,2,3,4,5,6,7,8,9,10,11,12,13,14,15);
}

__device__ __forceinline__ float epilogue_one(float accv, int col, float alpha,
                                              const float* bias, int actGelu) {
    float v = accv * alpha;
    if (bias)    v += bias[col];
    if (actGelu) v = 0.5f * v * (1.0f + erff(v * 0.70710678118f));
    return v;
}

// --------------------------- 64x64 variant ---------------------------
#define BK64   64
#define LDST64 72   // f16 per LDS row (64 data + 8 pad); 144B, 16B-aligned

__global__ __launch_bounds__(128)
void gemm_wmma64(const float* __restrict__ A, const float* __restrict__ Bm,
                 float* __restrict__ C, int M, int N, int K,
                 long long sA, long long sB, long long sC,
                 const float* __restrict__ bias,
                 const float* __restrict__ resid,
                 float alpha, int opB, int actGelu)
{
    __shared__ _Float16 shA[2][64 * LDST64];
    __shared__ _Float16 shB[2][64 * LDST64];   // stored [n][k]

    const int tid  = threadIdx.x;
    const int lane = tid & 31;
    const int wave = tid >> 5;
    const int m0   = blockIdx.y * 64;
    const int n0   = blockIdx.x * 64;
    const int bz   = blockIdx.z;

    A  += (long long)bz * sA;
    Bm += (long long)bz * sB;
    C  += (long long)bz * sC;
    const float* Rs = resid ? (resid + (long long)bz * sC) : (const float*)0;

    const int wr = (wave >> 1) * 32;
    const int wc = (wave & 1)  * 32;
    const int fm = lane & 15;
    const int fh = lane >> 4;

    v8f acc00 = {}, acc01 = {}, acc10 = {}, acc11 = {};
    float4 ra[8], rb[8];

    auto loadRegs = [&](int k0) {
#pragma unroll
        for (int it = 0; it < 8; ++it) {
            int e = it * 128 + tid;
            int row = e >> 4, c4 = (e & 15) * 4;
            ra[it] = *(const float4*)(A + (long long)(m0 + row) * K + k0 + c4);
        }
        if (opB) {
#pragma unroll
            for (int it = 0; it < 8; ++it) {
                int e = it * 128 + tid;
                int row = e >> 4, c4 = (e & 15) * 4;
                rb[it] = *(const float4*)(Bm + (long long)(n0 + row) * K + k0 + c4);
            }
        } else {
#pragma unroll
            for (int it = 0; it < 8; ++it) {
                int e = it * 128 + tid;
                int kr = e >> 4, n4 = (e & 15) * 4;
                rb[it] = *(const float4*)(Bm + (long long)(k0 + kr) * N + n0 + n4);
            }
        }
    };
    auto storeLds = [&](int buf) {
#pragma unroll
        for (int it = 0; it < 8; ++it) {
            int e = it * 128 + tid;
            int row = e >> 4, c4 = (e & 15) * 4;
            _Float16* d = &shA[buf][row * LDST64 + c4];
            d[0] = (_Float16)ra[it].x; d[1] = (_Float16)ra[it].y;
            d[2] = (_Float16)ra[it].z; d[3] = (_Float16)ra[it].w;
        }
        if (opB) {
#pragma unroll
            for (int it = 0; it < 8; ++it) {
                int e = it * 128 + tid;
                int row = e >> 4, c4 = (e & 15) * 4;
                _Float16* d = &shB[buf][row * LDST64 + c4];
                d[0] = (_Float16)rb[it].x; d[1] = (_Float16)rb[it].y;
                d[2] = (_Float16)rb[it].z; d[3] = (_Float16)rb[it].w;
            }
        } else {
#pragma unroll
            for (int it = 0; it < 8; ++it) {
                int e = it * 128 + tid;
                int kr = e >> 4, n4 = (e & 15) * 4;
                shB[buf][(n4 + 0) * LDST64 + kr] = (_Float16)rb[it].x;
                shB[buf][(n4 + 1) * LDST64 + kr] = (_Float16)rb[it].y;
                shB[buf][(n4 + 2) * LDST64 + kr] = (_Float16)rb[it].z;
                shB[buf][(n4 + 3) * LDST64 + kr] = (_Float16)rb[it].w;
            }
        }
    };
    auto compute = [&](int buf) {
#pragma unroll
        for (int kk = 0; kk < BK64; kk += 32) {
            v16h a0 = ldfragA(&shA[buf][(wr +      fm) * LDST64 + kk], fh);
            v16h a1 = ldfragA(&shA[buf][(wr + 16 + fm) * LDST64 + kk], fh);
            v16h b0 = ldfragB(&shB[buf][(wc +      fm) * LDST64 + kk], fh);
            v16h b1 = ldfragB(&shB[buf][(wc + 16 + fm) * LDST64 + kk], fh);
            acc00 = __builtin_amdgcn_wmma_f32_16x16x32_f16(false, a0, false, b0, (short)0, acc00, false, false);
            acc01 = __builtin_amdgcn_wmma_f32_16x16x32_f16(false, a0, false, b1, (short)0, acc01, false, false);
            acc10 = __builtin_amdgcn_wmma_f32_16x16x32_f16(false, a1, false, b0, (short)0, acc10, false, false);
            acc11 = __builtin_amdgcn_wmma_f32_16x16x32_f16(false, a1, false, b1, (short)0, acc11, false, false);
        }
    };

    loadRegs(0);
    storeLds(0);
    __syncthreads();
    int cur = 0;
    for (int k0 = 0; k0 < K; k0 += BK64) {
        const bool hasNext = (k0 + BK64) < K;
        if (hasNext) loadRegs(k0 + BK64);
        compute(cur);
        if (hasNext) {
            storeLds(cur ^ 1);
            __syncthreads();
            cur ^= 1;
        }
    }

    const int cn = lane & 15;
    const int cm = (lane >> 4) * 8;
    auto epi = [&](v8f acc, int ti, int tj) {
#pragma unroll
        for (int r = 0; r < 8; ++r) {
            int row = m0 + wr + ti * 16 + cm + r;
            int col = n0 + wc + tj * 16 + cn;
            float v = epilogue_one(acc[r], col, alpha, bias, actGelu);
            long long o = (long long)row * N + col;
            if (Rs) v += Rs[o];
            C[o] = v;
        }
    };
    epi(acc00, 0, 0); epi(acc01, 0, 1); epi(acc10, 1, 0); epi(acc11, 1, 1);
}

// --------------------------- 128x128 variant ---------------------------
#define BK128   32
#define LDST128 40   // f16 per LDS row (32 data + 8 pad); 80B, 16B-aligned

__global__ __launch_bounds__(256)
void gemm_wmma128(const float* __restrict__ A, const float* __restrict__ Bm,
                  float* __restrict__ C, int M, int N, int K,
                  long long sA, long long sB, long long sC,
                  const float* __restrict__ bias,
                  const float* __restrict__ resid,
                  float alpha, int opB, int actGelu)
{
    __shared__ _Float16 shA[2][128 * LDST128];
    __shared__ _Float16 shB[2][128 * LDST128];   // stored [n][k]

    const int tid  = threadIdx.x;
    const int lane = tid & 31;
    const int wave = tid >> 5;
    const int m0   = blockIdx.y * 128;
    const int n0   = blockIdx.x * 128;
    const int bz   = blockIdx.z;

    A  += (long long)bz * sA;
    Bm += (long long)bz * sB;
    C  += (long long)bz * sC;
    const float* Rs = resid ? (resid + (long long)bz * sC) : (const float*)0;

    const int wr = (wave >> 1) * 32;   // 0,32,64,96
    const int wc = (wave & 1)  * 64;   // 0,64
    const int fm = lane & 15;
    const int fh = lane >> 4;

    v8f acc[2][4] = {{{},{},{},{}},{{},{},{},{}}};
    float4 ra[4], rb[4];

    auto loadRegs = [&](int k0) {
#pragma unroll
        for (int it = 0; it < 4; ++it) {
            int e = it * 256 + tid;          // 0..1023
            int row = e >> 3;                // 128 rows, 8 float4 per row
            int c4  = (e & 7) * 4;
            ra[it] = *(const float4*)(A + (long long)(m0 + row) * K + k0 + c4);
        }
        if (opB) {
#pragma unroll
            for (int it = 0; it < 4; ++it) {
                int e = it * 256 + tid;
                int row = e >> 3, c4 = (e & 7) * 4;
                rb[it] = *(const float4*)(Bm + (long long)(n0 + row) * K + k0 + c4);
            }
        } else {
#pragma unroll
            for (int it = 0; it < 4; ++it) {
                int e = it * 256 + tid;
                int kr = e >> 5;             // 0..31 (k)
                int n4 = (e & 31) * 4;       // 0..124 (n)
                rb[it] = *(const float4*)(Bm + (long long)(k0 + kr) * N + n0 + n4);
            }
        }
    };
    auto storeLds = [&](int buf) {
#pragma unroll
        for (int it = 0; it < 4; ++it) {
            int e = it * 256 + tid;
            int row = e >> 3, c4 = (e & 7) * 4;
            _Float16* d = &shA[buf][row * LDST128 + c4];
            d[0] = (_Float16)ra[it].x; d[1] = (_Float16)ra[it].y;
            d[2] = (_Float16)ra[it].z; d[3] = (_Float16)ra[it].w;
        }
        if (opB) {
#pragma unroll
            for (int it = 0; it < 4; ++it) {
                int e = it * 256 + tid;
                int row = e >> 3, c4 = (e & 7) * 4;
                _Float16* d = &shB[buf][row * LDST128 + c4];
                d[0] = (_Float16)rb[it].x; d[1] = (_Float16)rb[it].y;
                d[2] = (_Float16)rb[it].z; d[3] = (_Float16)rb[it].w;
            }
        } else {
#pragma unroll
            for (int it = 0; it < 4; ++it) {
                int e = it * 256 + tid;
                int kr = e >> 5, n4 = (e & 31) * 4;
                shB[buf][(n4 + 0) * LDST128 + kr] = (_Float16)rb[it].x;
                shB[buf][(n4 + 1) * LDST128 + kr] = (_Float16)rb[it].y;
                shB[buf][(n4 + 2) * LDST128 + kr] = (_Float16)rb[it].z;
                shB[buf][(n4 + 3) * LDST128 + kr] = (_Float16)rb[it].w;
            }
        }
    };
    auto compute = [&](int buf) {
        v16h a0 = ldfragA(&shA[buf][(wr +      fm) * LDST128], fh);
        v16h a1 = ldfragA(&shA[buf][(wr + 16 + fm) * LDST128], fh);
#pragma unroll
        for (int j = 0; j < 4; ++j) {
            v16h b = ldfragB(&shB[buf][(wc + j * 16 + fm) * LDST128], fh);
            acc[0][j] = __builtin_amdgcn_wmma_f32_16x16x32_f16(false, a0, false, b, (short)0, acc[0][j], false, false);
            acc[1][j] = __builtin_amdgcn_wmma_f32_16x16x32_f16(false, a1, false, b, (short)0, acc[1][j], false, false);
        }
    };

    loadRegs(0);
    storeLds(0);
    __syncthreads();
    int cur = 0;
    for (int k0 = 0; k0 < K; k0 += BK128) {
        const bool hasNext = (k0 + BK128) < K;
        if (hasNext) loadRegs(k0 + BK128);
        compute(cur);
        if (hasNext) {
            storeLds(cur ^ 1);
            __syncthreads();
            cur ^= 1;
        }
    }

    const int cn = lane & 15;
    const int cm = (lane >> 4) * 8;
#pragma unroll
    for (int i = 0; i < 2; ++i)
#pragma unroll
        for (int j = 0; j < 4; ++j) {
#pragma unroll
            for (int r = 0; r < 8; ++r) {
                int row = m0 + wr + i * 16 + cm + r;
                int col = n0 + wc + j * 16 + cn;
                float v = epilogue_one(acc[i][j][r], col, alpha, bias, actGelu);
                long long o = (long long)row * N + col;
                if (Rs) v += Rs[o];
                C[o] = v;
            }
        }
}

// =====================================================================
// Elementwise / reduction kernels
// =====================================================================
__global__ __launch_bounds__(256)
void rmsnorm_k(const float* __restrict__ X, const float* __restrict__ g,
               float* __restrict__ Y, int d)
{
    __shared__ float red[256];
    long long row = blockIdx.x;
    const float* x = X + row * d;
    float*       y = Y + row * d;
    int t = threadIdx.x;
    float ss = 0.f;
    for (int j = t; j < d; j += 256) { float v = x[j]; ss += v * v; }
    red[t] = ss; __syncthreads();
    for (int o = 128; o > 0; o >>= 1) { if (t < o) red[t] += red[t + o]; __syncthreads(); }
    float norm = sqrtf(red[0]) * rsqrtf((float)d);
    float inv  = 1.0f / fmaxf(norm, 1e-8f);
    for (int j = t; j < d; j += 256) y[j] = x[j] * inv * g[j];
}

__global__ __launch_bounds__(256)
void softmax_k(float* __restrict__ S, int M, int Ncols, int causal)
{
    __shared__ float red[256];
    long long row = blockIdx.x;
    float* r = S + row * (long long)Ncols;
    int qi = (int)(row % M);
    int t = threadIdx.x;
    float mx = -3.0e38f;
    for (int j = t; j < Ncols; j += 256) {
        float v = r[j];
        if (causal && j > qi) v = -3.0e38f;
        mx = fmaxf(mx, v);
    }
    red[t] = mx; __syncthreads();
    for (int o = 128; o > 0; o >>= 1) { if (t < o) red[t] = fmaxf(red[t], red[t + o]); __syncthreads(); }
    mx = red[0]; __syncthreads();
    float sum = 0.f;
    for (int j = t; j < Ncols; j += 256) {
        float v = (causal && j > qi) ? 0.f : __expf(r[j] - mx);
        r[j] = v; sum += v;
    }
    red[t] = sum; __syncthreads();
    for (int o = 128; o > 0; o >>= 1) { if (t < o) red[t] += red[t + o]; __syncthreads(); }
    float is = 1.0f / red[0];
    for (int j = t; j < Ncols; j += 256) r[j] *= is;
}

__global__ __launch_bounds__(256)
void embed_k(const int* __restrict__ tok, const float* __restrict__ emb,
             const float* __restrict__ pos, float* __restrict__ out,
             long long total, int dim, int nper)
{
    long long idx = (long long)blockIdx.x * 256 + threadIdx.x;
    if (idx >= total) return;
    int c = (int)(idx % dim);
    long long row = idx / dim;
    int t = tok[row];
    float v = emb[(long long)t * dim + c];
    if (pos) v += pos[(long long)(row % nper) * dim + c];
    out[idx] = v;
}

__global__ __launch_bounds__(256)
void ectx_k(const float* __restrict__ embed, float* __restrict__ out, long long total)
{
    long long idx = (long long)blockIdx.x * 256 + threadIdx.x;
    if (idx >= total) return;
    int c = (int)(idx % ENCD);
    long long row = idx / ENCD;
    int m = (int)(row % CNN);
    int k = (int)((row / (CNN * RR)) % KCH);
    int b = (int)(row / (CNN * RR * KCH));
    out[idx] = embed[(long long)(b * NN + k * CNN + m) * ENCD + c];
}

// [nb, n, ld] slice (colOff..colOff+512) -> [nb*8, n, 64] with optional RoPE
__global__ __launch_bounds__(256)
void split_heads_k(const float* __restrict__ X, float* __restrict__ O,
                   int nb, int n, int ld, int colOff, int rope, int posOff)
{
    int idx = blockIdx.x * 256 + threadIdx.x;
    int total = nb * n * INNER;
    if (idx >= total) return;
    int d  = idx & 63;
    int h  = (idx >> 6) & 7;
    int ni = (idx >> 9) % n;
    int bi = idx / (n << 9);
    const float* xr = X + (long long)(bi * n + ni) * ld + colOff + h * 64;
    float v = xr[d];
    if (rope) {
        int i = d & 31;
        float invf = __powf(10000.0f, -(float)(2 * i) * (1.0f / 64.0f));
        float f = (float)(ni + posOff) * invf;
        float o = (d < 32) ? -xr[d + 32] : xr[d - 32];
        v = v * __cosf(f) + o * __sinf(f);
    }
    O[((long long)(bi * NHEADS + h) * n + ni) * 64 + d] = v;
}

// [nb*8, n, 64] -> [nb*n, 512]
__global__ __launch_bounds__(256)
void merge_heads_k(const float* __restrict__ I, float* __restrict__ O, int nb, int n)
{
    int idx = blockIdx.x * 256 + threadIdx.x;
    int total = nb * n * INNER;
    if (idx >= total) return;
    int d  = idx & 63;
    int h  = (idx >> 6) & 7;
    int ni = (idx >> 9) % n;
    int bi = idx / (n << 9);
    O[(long long)(bi * n + ni) * INNER + h * 64 + d] =
        I[((long long)(bi * NHEADS + h) * n + ni) * 64 + d];
}

// Y[b,i,:] = (i+cp < N) ? X[b,i+cp,:] : 0      (cp = CN-1)
__global__ __launch_bounds__(256)
void shift_k(const float* __restrict__ X, float* __restrict__ Y, long long total)
{
    long long idx = (long long)blockIdx.x * 256 + threadIdx.x;
    if (idx >= total) return;
    int c = (int)(idx % DECD);
    int i = (int)((idx / DECD) % NN);
    int b = (int)(idx / ((long long)DECD * NN));
    int src = i + (CNN - 1);
    Y[idx] = (src < NN) ? X[(long long)(b * NN + src) * DECD + c] : 0.f;
}

// X[b,i,:] += (i >= cp) ? A[b,i-cp,:] : 0
__global__ __launch_bounds__(256)
void unshift_add_k(float* __restrict__ X, const float* __restrict__ A, long long total)
{
    long long idx = (long long)blockIdx.x * 256 + threadIdx.x;
    if (idx >= total) return;
    int c = (int)(idx % DECD);
    int i = (int)((idx / DECD) % NN);
    int b = (int)(idx / ((long long)DECD * NN));
    int src = i - (CNN - 1);
    if (src >= 0) X[idx] += A[(long long)(b * NN + src) * DECD + c];
}

// =====================================================================
// Host-side orchestration
// =====================================================================
struct AttnP { const float *gamma, *wq, *wkv, *wo, *bo; };
struct FFP   { const float *gamma, *w1, *b1, *w2, *b2; };
struct LayerP { AttnP attn; int hasCross; AttnP cross; FFP ff; };

struct Ws {
    float *xn, *q, *kv, *qh, *kh, *vh, *sim, *oh, *merged, *ffh;
};

static inline void gemm(hipStream_t s,
    const float* A, const float* Bm, float* C,
    int M, int N, int K, int batch,
    long long sA, long long sB, long long sC,
    const float* bias, const float* resid,
    float alpha, int opB, int act)
{
    if ((M % 128) == 0 && (N % 128) == 0) {
        dim3 grid(N / 128, M / 128, batch);
        gemm_wmma128<<<grid, 256, 0, s>>>(A, Bm, C, M, N, K, sA, sB, sC, bias, resid, alpha, opB, act);
    } else {
        dim3 grid(N / 64, M / 64, batch);
        gemm_wmma64<<<grid, 128, 0, s>>>(A, Bm, C, M, N, K, sA, sB, sC, bias, resid, alpha, opB, act);
    }
}

static inline int gsz(long long total) { return (int)((total + 255) / 256); }

static void attention_fwd(hipStream_t s, const Ws& w,
                          const float* x, const float* ctx,
                          int nb, int qn, int kvn, int dim, const AttnP& P,
                          int causal, int qOff, int kOff,
                          float* outC, const float* resid)
{
    int qrows = nb * qn, krows = nb * kvn;
    rmsnorm_k<<<qrows, 256, 0, s>>>(x, P.gamma, w.xn, dim);
    gemm(s, w.xn, P.wq, w.q, qrows, INNER, dim, 1, 0, 0, 0, nullptr, nullptr, 1.f, 0, 0);
    const float* kvin = ctx ? ctx : x;
    gemm(s, kvin, P.wkv, w.kv, krows, 2 * INNER, dim, 1, 0, 0, 0, nullptr, nullptr, 1.f, 0, 0);

    long long tq = (long long)qrows * INNER, tk = (long long)krows * INNER;
    split_heads_k<<<gsz(tq), 256, 0, s>>>(w.q,  w.qh, nb, qn,  INNER,     0,     1, qOff);
    split_heads_k<<<gsz(tk), 256, 0, s>>>(w.kv, w.kh, nb, kvn, 2 * INNER, 0,     1, kOff);
    split_heads_k<<<gsz(tk), 256, 0, s>>>(w.kv, w.vh, nb, kvn, 2 * INNER, INNER, 0, 0);

    // sim = (q * d^-1/2) @ k^T  : batched, B row-major [kvn, 64] -> opB=1
    gemm(s, w.qh, w.kh, w.sim, qn, kvn, DHEAD, nb * NHEADS,
         (long long)qn * 64, (long long)kvn * 64, (long long)qn * kvn,
         nullptr, nullptr, 0.125f, 1, 0);
    softmax_k<<<nb * NHEADS * qn, 256, 0, s>>>(w.sim, qn, kvn, causal);
    // out = attn @ v : v is [kvn, 64] row-major K x N -> opB=0
    gemm(s, w.sim, w.vh, w.oh, qn, DHEAD, kvn, nb * NHEADS,
         (long long)qn * kvn, (long long)kvn * 64, (long long)qn * 64,
         nullptr, nullptr, 1.f, 0, 0);
    merge_heads_k<<<gsz(tq), 256, 0, s>>>(w.oh, w.merged, nb, qn);
    gemm(s, w.merged, P.wo, outC, qrows, dim, INNER, 1, 0, 0, 0, P.bo, resid, 1.f, 0, 0);
}

static void ff_fwd(hipStream_t s, const Ws& w, float* x, int rows, int dim, const FFP& P)
{
    rmsnorm_k<<<rows, 256, 0, s>>>(x, P.gamma, w.xn, dim);
    gemm(s, w.xn, P.w1, w.ffh, rows, 4 * dim, dim, 1, 0, 0, 0, P.b1, nullptr, 1.f, 0, 1);
    gemm(s, w.ffh, P.w2, x, rows, dim, 4 * dim, 1, 0, 0, 0, P.b2, x, 1.f, 0, 0);
}

extern "C" void kernel_launch(void* const* d_in, const int* in_sizes, int n_in,
                              void* d_out, int out_size, void* d_ws, size_t ws_size,
                              hipStream_t stream)
{
    (void)in_sizes; (void)n_in; (void)out_size; (void)ws_size;
    const int* seq       = (const int*)d_in[0];
    const int* retrieved = (const int*)d_in[1];

    // ---- params in jax pytree leaf order (sorted dict keys, None dropped) ----
    int pi = 2;
    auto F = [&]() -> const float* { return (const float*)d_in[pi++]; };
    auto readAttn = [&](AttnP& a) { a.bo = F(); a.gamma = F(); a.wkv = F(); a.wo = F(); a.wq = F(); };
    auto readFF   = [&](FFP& f)   { f.b1 = F(); f.b2 = F(); f.gamma = F(); f.w1 = F(); f.w2 = F(); };

    const float* dec_gamma = F();
    LayerP dec[12];
    for (int l = 0; l < 12; ++l) {
        readAttn(dec[l].attn);
        dec[l].hasCross = (l == 0 || l == 2 || l == 5 || l == 8);
        if (dec[l].hasCross) readAttn(dec[l].cross);
        readFF(dec[l].ff);
    }
    const float* enc2dec_b = F();
    const float* enc2dec_w = F();
    const float* enc_gamma = F();
    LayerP enc[2];
    for (int l = 0; l < 2; ++l) {
        readAttn(enc[l].attn);
        enc[l].hasCross = 1;
        readAttn(enc[l].cross);
        readFF(enc[l].ff);
    }
    const float* logit_b   = F();
    const float* logit_w   = F();
    const float* pos_emb   = F();
    const float* to_dec_b  = F();
    const float* to_dec_w  = F();
    const float* token_emb = F();

    // ---- carve workspace ----
    char* wp = (char*)d_ws;
    auto alloc = [&](long long n) -> float* {
        float* r = (float*)wp;
        wp += ((n * 4 + 255) / 256) * 256;
        return r;
    };
    Ws w;
    w.xn     = alloc(2048LL * ENCD);
    w.q      = alloc(2048LL * INNER);
    w.kv     = alloc(2048LL * 2 * INNER);
    w.qh     = alloc(2048LL * INNER);
    w.kh     = alloc(2048LL * INNER);
    w.vh     = alloc(2048LL * INNER);
    w.sim    = alloc(16LL * 512 * 512);
    w.oh     = alloc(2048LL * INNER);
    w.merged = alloc(2048LL * INNER);
    w.ffh    = alloc(2048LL * 4 * ENCD);
    float* embed   = alloc((long long)BB * NN * ENCD);     // 1024 x 896
    float* retr    = alloc(2048LL * ENCD);                 // 2048 x 896
    float* ectx    = alloc(2048LL * ENCD);
    float* encN    = alloc(2048LL * ENCD);
    float* retrDec = alloc(2048LL * DECD);                 // 2048 x 768
    float* xdec    = alloc((long long)BB * NN * DECD);     // 1024 x 768
    float* x2      = alloc((long long)BB * NN * DECD);
    float* cOut    = alloc((long long)BB * NN * DECD);

    // ---- embeddings ----
    long long tEmb  = (long long)BB * NN * ENCD;
    long long tRetr = 2048LL * ENCD;
    embed_k<<<gsz(tEmb),  256, 0, stream>>>(seq,       token_emb, pos_emb, embed, tEmb,  ENCD, NN);
    embed_k<<<gsz(tRetr), 256, 0, stream>>>(retrieved, token_emb, nullptr, retr,  tRetr, ENCD, CNN);
    ectx_k<<<gsz(tRetr), 256, 0, stream>>>(embed, ectx, tRetr);

    // ---- encoder: 32 sequences of len 64, dim 896 ----
    for (int l = 0; l < 2; ++l) {
        attention_fwd(stream, w, retr, nullptr, 32, 64, 64, ENCD, enc[l].attn, 0, 0, 0, retr, retr);
        attention_fwd(stream, w, retr, ectx,    32, 64, 64, ENCD, enc[l].cross, 0, 0, 0, retr, retr);
        ff_fwd(stream, w, retr, 2048, ENCD, enc[l].ff);
    }
    rmsnorm_k<<<2048, 256, 0, stream>>>(retr, enc_gamma, encN, ENCD);
    // enc2dec projection: 2048 x 896 -> 2048 x 768
    gemm(stream, encN, enc2dec_w, retrDec, 2048, DECD, ENCD, 1, 0, 0, 0, enc2dec_b, nullptr, 1.f, 0, 0);
    // to_dec projection: 1024 x 896 -> 1024 x 768
    gemm(stream, embed, to_dec_w, xdec, BB * NN, DECD, ENCD, 1, 0, 0, 0, to_dec_b, nullptr, 1.f, 0, 0);

    // ---- decoder: 2 sequences of len 512, dim 768 ----
    long long tX = (long long)BB * NN * DECD;
    for (int l = 0; l < 12; ++l) {
        attention_fwd(stream, w, xdec, nullptr, BB, NN, NN, DECD, dec[l].attn, 1, 0, 0, xdec, xdec);
        if (dec[l].hasCross) {
            shift_k<<<gsz(tX), 256, 0, stream>>>(xdec, x2, tX);
            // chunked: 16 sequences, q len 64 (rope offset 63), kv = retrDec [16,128,768]
            attention_fwd(stream, w, x2, retrDec, BB * KCH, NN / KCH, RR * CNN, DECD,
                          dec[l].cross, 0, CNN - 1, 0, cOut, nullptr);
            unshift_add_k<<<gsz(tX), 256, 0, stream>>>(xdec, cOut, tX);
        }
        ff_fwd(stream, w, xdec, BB * NN, DECD, dec[l].ff);
    }
    rmsnorm_k<<<BB * NN, 256, 0, stream>>>(xdec, dec_gamma, w.xn, DECD);

    // ---- logits: 1024 x 768 -> 1024 x 32000 into d_out ----
    gemm(stream, w.xn, logit_w, (float*)d_out, BB * NN, VOCAB, DECD, 1, 0, 0, 0,
         logit_b, nullptr, 1.f, 0, 0);
}
